// MultiHeadAttention_81037442941414
// MI455X (gfx1250) — matrix operand order repaired and checked by reading, not compiled, not run on hardware
//
#include <hip/hip_runtime.h>
#include <hip/hip_bf16.h>

// ---------------------------------------------------------------------------
// MultiHeadAttention for MI455X (gfx1250, wave32, WMMA bf16 16x16x32).
//
// ~38 GFLOP vs ~6MB true I/O -> matrix-core bound. bf16 WMMA with f32
// accumulation; flash-attention (2048x2048 scores never touch memory).
// K/V chunks double-buffered in LDS via GLOBAL_LOAD_ASYNC_TO_LDS_B128
// (ASYNCcnt), so the copy latency hides under the 32 WMMAs per chunk.
// ---------------------------------------------------------------------------

typedef __attribute__((ext_vector_type(16))) __bf16 v16bf;
typedef __attribute__((ext_vector_type(8)))  float  v8f;

#define N_EMBD  128
#define N_HEADS 8
#define BATCH   2
#define SEQ     2048
#define INNER   1024
#define ROWS    (BATCH * SEQ)          // 4096 merged rows
#define NCH     (SEQ / 64)             // key chunks per row-block

#define USE_ASYNC_LDS 1

// ---- CDNA5 async global->LDS copy (16B per lane), tracked by ASYNCcnt ------
__device__ __forceinline__ void async_copy_b128(void* lds, const void* gsrc) {
#if USE_ASYNC_LDS
    // Generic shared pointer: low 32 bits are the wave-relative LDS address
    // (ISA 10.2 aperture mapping), which is what VDST of the async op takes.
    unsigned l = (unsigned)(uintptr_t)lds;
    asm volatile("global_load_async_to_lds_b128 %0, %1, off"
                 :: "v"(l), "v"(gsrc) : "memory");
#else
    *reinterpret_cast<uint4*>(lds) = *reinterpret_cast<const uint4*>(gsrc);
#endif
}
__device__ __forceinline__ void async_wait_all() {
#if USE_ASYNC_LDS
    asm volatile("s_wait_asynccnt 0x0" ::: "memory");
#endif
}

// ---- WMMA fragment loaders -------------------------------------------------
// A operand: 16x32 (MxK). Lane: m=lane&15, g=lane>>4 holds K = {8g..8g+7} and
// {16+8g..16+8g+7} (ISA 7.12.2 "16-bit A-Matrix 16x32").
__device__ __forceinline__ v16bf ldA_bf16(const __bf16* src, int stride) {
    const int lane = threadIdx.x & 31;
    const int m = lane & 15, g = lane >> 4;
    const __bf16* p = src + m * stride + g * 8;
    v16bf f;
#pragma unroll
    for (int i = 0; i < 8; ++i) { f[i] = p[i]; f[8 + i] = p[16 + i]; }
    return f;
}
// B operand: 32x16 (KxN), loaded from an N-major (transposed) source:
// lane n=lane&15 holds K = {8g..8g+7} and {16+8g..16+8g+7} of column n.
__device__ __forceinline__ v16bf ldB_bf16(const __bf16* srcT, int stride) {
    const int lane = threadIdx.x & 31;
    const int n = lane & 15, g = lane >> 4;
    const __bf16* p = srcT + n * stride + g * 8;
    v16bf f;
#pragma unroll
    for (int i = 0; i < 8; ++i) { f[i] = p[i]; f[8 + i] = p[16 + i]; }
    return f;
}

__device__ __forceinline__ v8f wmma_bf16(v16bf a, v16bf b, v8f c) {
    return __builtin_amdgcn_wmma_f32_16x16x32_bf16(
        false, a, false, b, (short)0, c, false, false);
}

// ---------------------------------------------------------------------------
// Kernel 0: transpose + convert weights to bf16, pack biases, convert x.
//   WTqkv[mat][out][in] (3x1024x128 bf16), WpT[out][in] (128x1024 bf16),
//   bqkv[mat][out] (3x1024 f32), xb = bf16(x) (4096x128).
// ---------------------------------------------------------------------------
__global__ __launch_bounds__(256) void prep_kernel(
    const float* __restrict__ x,
    const float* __restrict__ Wk, const float* __restrict__ bk,
    const float* __restrict__ Wq, const float* __restrict__ bq,
    const float* __restrict__ Wv, const float* __restrict__ bv,
    const float* __restrict__ Wp,
    __bf16* __restrict__ WTqkv, __bf16* __restrict__ WpT,
    float* __restrict__ bqkv, __bf16* __restrict__ xb)
{
    const int idx = blockIdx.x * 256 + threadIdx.x;
    const int QKVW = 3 * INNER * N_EMBD;                 // 393216
    const int WPE  = QKVW + N_EMBD * INNER;              // 524288
    const int BE   = WPE + 3 * INNER;                    // 527360
    const int XE   = BE + ROWS * N_EMBD;                 // 1051648
    if (idx < QKVW) {
        const int mat = idx / (INNER * N_EMBD);
        const int r   = idx % (INNER * N_EMBD);
        const int o = r >> 7, i = r & 127;
        const float* W = (mat == 0) ? Wk : ((mat == 1) ? Wq : Wv);
        WTqkv[idx] = (__bf16)W[i * INNER + o];
    } else if (idx < WPE) {
        const int r = idx - QKVW;
        const int o = r >> 10, i = r & 1023;
        WpT[r] = (__bf16)Wp[i * N_EMBD + o];
    } else if (idx < BE) {
        const int r = idx - WPE;
        const int mat = r >> 10, o = r & 1023;
        const float* bsrc = (mat == 0) ? bk : ((mat == 1) ? bq : bv);
        bqkv[r] = bsrc[o];
    } else if (idx < XE) {
        const int r = idx - BE;
        xb[r] = (__bf16)x[r];
    }
}

// ---------------------------------------------------------------------------
// Kernel 1: QKV projection.  y = x @ W + b, bf16 outputs:
//   mat 0 -> K head-major [b][h][l][128]
//   mat 1 -> Q head-major [b][h][l][128]
//   mat 2 -> V^T          [b][h][128][l]   (one b128 store per lane!)
// Each wave: one 16x16 tile, K=128 -> 4 WMMAs.
// ---------------------------------------------------------------------------
__global__ __launch_bounds__(256) void qkv_kernel(
    const __bf16* __restrict__ xb, const __bf16* __restrict__ WTqkv,
    const float* __restrict__ bqkv, __bf16* __restrict__ KQV)
{
    const int mtile = blockIdx.x;                   // 0..255 (16-row tiles)
    const int wave  = threadIdx.x >> 5;
    const int nt    = blockIdx.y * 8 + wave;        // 0..191 (3 mats x 64)
    const int mat   = nt >> 6;
    const int col0  = (nt & 63) * 16;
    const int lane  = threadIdx.x & 31;
    const int g = lane >> 4, cn = lane & 15;
    const int row0 = mtile * 16;

    const __bf16* xa = xb + (size_t)row0 * N_EMBD;
    const __bf16* WT = WTqkv + (size_t)mat * INNER * N_EMBD;

    v8f acc = {};
#pragma unroll
    for (int kk = 0; kk < 4; ++kk) {
        v16bf a = ldA_bf16(xa + kk * 32, N_EMBD);
        v16bf b = ldB_bf16(WT + (size_t)col0 * N_EMBD + kk * 32, N_EMBD);
        acc = wmma_bf16(a, b, acc);
    }
    const float bias = bqkv[mat * INNER + col0 + cn];
    const int h = col0 >> 7;
    const int d = (col0 & 127) + cn;

    if (mat == 2) {
        // V^T: lane's 8 values are 8 consecutive seq positions -> 1 x b128
        __bf16 tmp[8];
#pragma unroll
        for (int r = 0; r < 8; ++r) tmp[r] = (__bf16)(acc[r] + bias);
        const int row = row0 + 8 * g;
        const int bb = row >> 11, l0 = row & (SEQ - 1);
        __bf16* Vt = KQV + 2 * (size_t)ROWS * INNER;
        *reinterpret_cast<uint4*>(
            Vt + (((size_t)bb * N_HEADS + h) * N_EMBD + d) * SEQ + l0) =
            *reinterpret_cast<const uint4*>(tmp);
    } else {
        __bf16* out = KQV + (size_t)mat * ROWS * INNER;
#pragma unroll
        for (int r = 0; r < 8; ++r) {
            const int row = row0 + r + 8 * g;
            const int bb = row >> 11, l = row & (SEQ - 1);
            out[(((size_t)bb * N_HEADS + h) * SEQ + l) * N_EMBD + d] =
                (__bf16)(acc[r] + bias);
        }
    }
}

// ---------------------------------------------------------------------------
// Kernel 2: flash attention, async double-buffered K/V staging.
// Block = 256 thr (8 waves) handles 128 query rows of one (b,h); wave = 16
// queries. S^T = K_tile x Q puts softmax rows in lanes; P^T's C-layout equals
// the A-operand layout for P x V (register-only fragment build).
// ---------------------------------------------------------------------------
__global__ __launch_bounds__(256) void attn_kernel(
    const __bf16* __restrict__ KQV, __bf16* __restrict__ Om)
{
    const __bf16* Khm = KQV;                              // [bh][l][128]
    const __bf16* Qhm = KQV + (size_t)ROWS * INNER;       // [bh][l][128]
    const __bf16* Vtg = KQV + 2 * (size_t)ROWS * INNER;   // [bh][128][l]

    __shared__ __align__(16) __bf16 Ks[2][64][128];       // 32 KB
    __shared__ __align__(16) __bf16 Vst[2][128][64];      // 32 KB (V^T chunk)

    const int bh   = blockIdx.x >> 4;
    const int qblk = blockIdx.x & 15;
    const int tid  = threadIdx.x;
    const int wave = tid >> 5, lane = tid & 31;
    const int g = lane >> 4, qn = lane & 15;
    const int q0 = qblk * 128 + wave * 16;

    const __bf16* Kbh = Khm + (size_t)bh * SEQ * N_EMBD;
    const __bf16* Vbh = Vtg + (size_t)bh * N_EMBD * SEQ;

    // cooperative async staging of chunk j into buffer `buf`
    auto stage = [&](int buf, int j) {
#pragma unroll
        for (int it = 0; it < 4; ++it) {
            const int e = tid + it * 256;                 // 1024 x 16B each
            const int key = e >> 4, off = (e & 15) * 8;   // K: [64][128]
            async_copy_b128(&Ks[buf][key][off],
                            Kbh + (size_t)(j * 64 + key) * N_EMBD + off);
            const int d = e >> 3, ko = (e & 7) * 8;       // V^T: [128][64]
            async_copy_b128(&Vst[buf][d][ko],
                            Vbh + (size_t)d * SEQ + j * 64 + ko);
        }
    };

    stage(0, 0);

    // Q fragments (B-operand of S^T = K x Q): row q of Q = column q of Q^T
    const __bf16* Qbase = Qhm + ((size_t)bh * SEQ + q0) * N_EMBD;
    v16bf qf[4];
#pragma unroll
    for (int kk = 0; kk < 4; ++kk) qf[kk] = ldB_bf16(Qbase + kk * 32, N_EMBD);

    v8f o[8];
#pragma unroll
    for (int n = 0; n < 8; ++n) o[n] = (v8f){};
    float mstate = -1e30f, lstate = 0.0f;
    const float scale = 0.08838834764831845f;             // 1/sqrt(128)

    async_wait_all();
    __syncthreads();

    for (int j = 0; j < NCH; ++j) {
        const int cur = j & 1;
        if (j + 1 < NCH) stage(cur ^ 1, j + 1);           // prefetch next chunk

        // S^T tiles: 4 x (16 keys x 16 queries), K=128 -> 4 WMMAs each
        v8f st[4];
#pragma unroll
        for (int t = 0; t < 4; ++t) {
            v8f s = (v8f){};
#pragma unroll
            for (int kk = 0; kk < 4; ++kk) {
                v16bf ka = ldA_bf16(&Ks[cur][t * 16][0] + kk * 32, 128);
                s = wmma_bf16(ka, qf[kk], s);
            }
#pragma unroll
            for (int r = 0; r < 8; ++r) s[r] *= scale;
            st[t] = s;
        }

        // Online softmax for query qn (lane pair {qn, qn+16} covers the keys)
        float cmax = -1e30f;
#pragma unroll
        for (int t = 0; t < 4; ++t)
#pragma unroll
            for (int r = 0; r < 8; ++r) cmax = fmaxf(cmax, st[t][r]);
        cmax = fmaxf(cmax, __shfl_xor(cmax, 16, 32));
        const float mnew = fmaxf(mstate, cmax);
        float lsum = 0.0f;
#pragma unroll
        for (int t = 0; t < 4; ++t)
#pragma unroll
            for (int r = 0; r < 8; ++r) {
                const float p = __expf(st[t][r] - mnew);
                st[t][r] = p;
                lsum += p;
            }
        lsum += __shfl_xor(lsum, 16, 32);
        const float alpha = __expf(mstate - mnew);
        lstate = lstate * alpha + lsum;
        mstate = mnew;

        // Rescale O (row r+8g is query r+8g -> bpermute its alpha)
        float ar[8];
#pragma unroll
        for (int r = 0; r < 8; ++r) ar[r] = __shfl(alpha, r + 8 * g, 32);
#pragma unroll
        for (int n = 0; n < 8; ++n)
#pragma unroll
            for (int r = 0; r < 8; ++r) o[n][r] *= ar[r];

        // P fragments: P^T C-layout == A-operand layout (register-only)
        v16bf pf0, pf1;
#pragma unroll
        for (int i = 0; i < 8; ++i) {
            pf0[i]     = (__bf16)st[0][i];
            pf0[8 + i] = (__bf16)st[1][i];
            pf1[i]     = (__bf16)st[2][i];
            pf1[8 + i] = (__bf16)st[3][i];
        }

        // O += P x V  (8 dim-tiles x 2 key-slices)
#pragma unroll
        for (int n = 0; n < 8; ++n) {
            v16bf vb0 = ldB_bf16(&Vst[cur][n * 16][0] + 0,  64);
            v16bf vb1 = ldB_bf16(&Vst[cur][n * 16][0] + 32, 64);
            o[n] = wmma_bf16(pf0, vb0, o[n]);
            o[n] = wmma_bf16(pf1, vb1, o[n]);
        }

        if (j + 1 < NCH) { async_wait_all(); __syncthreads(); }
    }

    // Normalize and store merged-head bf16 [B][L][1024]
    float lr[8];
#pragma unroll
    for (int r = 0; r < 8; ++r) lr[r] = __shfl(lstate, r + 8 * g, 32);
    const int b = bh >> 3, h = bh & 7;
#pragma unroll
    for (int r = 0; r < 8; ++r) {
        const float inv = 1.0f / lr[r];
        const int l = q0 + r + 8 * g;
#pragma unroll
        for (int n = 0; n < 8; ++n) {
            Om[((size_t)b * SEQ + l) * INNER + h * N_EMBD + n * 16 + qn] =
                (__bf16)(o[n][r] * inv);
        }
    }
}

// ---------------------------------------------------------------------------
// Kernel 3: output projection  y = Om @ Wp + bp  (f32 out). K=1024.
// ---------------------------------------------------------------------------
__global__ __launch_bounds__(256) void proj_kernel(
    const __bf16* __restrict__ Om, const __bf16* __restrict__ WpT,
    const float* __restrict__ bp, float* __restrict__ out)
{
    const int mtile = blockIdx.x;                   // 0..255
    const int wave  = threadIdx.x >> 5;             // column tile 0..7
    const int lane  = threadIdx.x & 31;
    const int g = lane >> 4, cn = lane & 15;
    const int row0 = mtile * 16, col0 = wave * 16;

    const __bf16* A = Om + (size_t)row0 * INNER;
    v8f acc = {};
#pragma unroll 4
    for (int kk = 0; kk < 32; ++kk) {
        v16bf a = ldA_bf16(A + kk * 32, INNER);
        v16bf b = ldB_bf16(WpT + (size_t)col0 * INNER + kk * 32, INNER);
        acc = wmma_bf16(a, b, acc);
    }
    const float bias = bp[col0 + cn];
#pragma unroll
    for (int r = 0; r < 8; ++r)
        out[(size_t)(row0 + r + 8 * g) * N_EMBD + col0 + cn] = acc[r] + bias;
}

// ---------------------------------------------------------------------------
extern "C" void kernel_launch(void* const* d_in, const int* in_sizes, int n_in,
                              void* d_out, int out_size, void* d_ws, size_t ws_size,
                              hipStream_t stream) {
    (void)in_sizes; (void)n_in; (void)out_size; (void)ws_size;
    const float* x  = (const float*)d_in[0];
    const float* Wk = (const float*)d_in[1];
    const float* bk = (const float*)d_in[2];
    const float* Wq = (const float*)d_in[3];
    const float* bq = (const float*)d_in[4];
    const float* Wv = (const float*)d_in[5];
    const float* bv = (const float*)d_in[6];
    const float* Wp = (const float*)d_in[7];
    const float* bp = (const float*)d_in[8];

    // Workspace layout (bytes), total ~35.7 MB:
    char* ws = (char*)d_ws;
    __bf16* WTqkv = (__bf16*)(ws);                        //    786,432 B
    __bf16* WpT   = (__bf16*)(ws + 786432);               //    262,144 B
    float*  bqkv  = (float* )(ws + 1048576);              //     12,288 B
    __bf16* KQV   = (__bf16*)(ws + 1060864);              // 25,165,824 B (K,Q head-major; V^T)
    __bf16* Om    = (__bf16*)(ws + 1060864 + 25165824);   //  8,388,608 B
    __bf16* xb    = (__bf16*)(ws + 1060864 + 25165824 + 8388608); // 1,048,576 B

    prep_kernel<<<4108, 256, 0, stream>>>(x, Wk, bk, Wq, bq, Wv, bv, Wp,
                                          WTqkv, WpT, bqkv, xb);
    qkv_kernel<<<dim3(256, 24), 256, 0, stream>>>(xb, WTqkv, bqkv, KQV);
    attn_kernel<<<BATCH * N_HEADS * (SEQ / 128), 256, 0, stream>>>(KQV, Om);
    proj_kernel<<<256, 256, 0, stream>>>(Om, WpT, bp, (float*)d_out);
}
